// DeformConv_88373247083199
// MI455X (gfx1250) — matrix operand order repaired and checked by reading, not compiled, not run on hardware
//
#include <hip/hip_runtime.h>
#include <stdint.h>

typedef __attribute__((ext_vector_type(16))) __bf16 v16bf;
typedef __attribute__((ext_vector_type(8)))  __bf16 v8bf;
typedef __attribute__((ext_vector_type(2)))  __bf16 v2bf;
typedef __attribute__((ext_vector_type(8)))  float  v8f;
typedef __attribute__((ext_vector_type(4)))  int    v4i;

#define Bn   2
#define Ci   256
#define Cco  256
#define Hd   96
#define Wd   96
#define HW   9216          // Hd*Wd
#define KT   2304          // Ci*9
#define NPIX 18432         // Bn*HW
#define PIXT 288           // number of 8x8 pixel tiles
#define LDT  40            // padded LDS row length in bf16 elements (80B)

// workspace layout (bytes)
#define WS_WB   0u           // 256*2304 bf16  = 1,179,648
#define WS_WO   1179648u     // 32*2304  bf16  =   147,456
#define WS_OM   1327104u     // 2*27*9216 f32  = 1,990,656
#define WS_PRE  3317760u     // 2*256*9216 f32 = 18,874,368
#define WS_PS   22192128u    // 256*288 f32    =   294,912
#define WS_PQ   22487040u    // 256*288 f32    =   294,912
#define WS_ST   22781952u    // 512 f32

#if __has_builtin(__builtin_amdgcn_global_load_async_to_lds_b128)
#define HAVE_ASYNC 1
#else
#define HAVE_ASYNC 0
#endif

// 16B global -> LDS copy; async (ASYNCcnt) when the toolchain exposes it.
__device__ __forceinline__ void cp16(const __bf16* g, __bf16* l) {
#if HAVE_ASYNC
  __builtin_amdgcn_global_load_async_to_lds_b128(
      (__attribute__((address_space(1))) v4i*)g,
      (__attribute__((address_space(3))) v4i*)l, 0, 0);
#else
  *(uint4*)l = *(const uint4*)g;
#endif
}

__device__ __forceinline__ void async_wait() {
#if HAVE_ASYNC
#if __has_builtin(__builtin_amdgcn_s_wait_asynccnt)
  __builtin_amdgcn_s_wait_asynccnt(0);
#else
  asm volatile("s_wait_asynccnt 0" ::: "memory");
#endif
#endif
}

__device__ __forceinline__ unsigned pack2bf(float a, float b) {
#if __has_builtin(__builtin_amdgcn_cvt_pk_bf16_f32)
  v2bf r = __builtin_amdgcn_cvt_pk_bf16_f32(a, b);
  union { v2bf v; unsigned u; } u; u.v = r; return u.u;
#else
  union { __bf16 h[2]; unsigned u; } u;
  u.h[0] = (__bf16)a; u.h[1] = (__bf16)b;
  return u.u;
#endif
}

__device__ __forceinline__ v8f wmma_bf16(v16bf a, v16bf b, v8f c) {
  return __builtin_amdgcn_wmma_f32_16x16x32_bf16(false, a, false, b, (short)0, c,
                                                 false, false);
}

// A fragment (16x32 bf16, M=rows in LDS, K contiguous per row, row stride LDT)
__device__ __forceinline__ v16bf load_a(const __bf16* A, int m0, int lane) {
  int row = m0 + (lane & 15);
  const __bf16* p = A + row * LDT + ((lane & 16) ? 8 : 0);
  union { v16bf v; v8bf h[2]; } u;
  u.h[0] = *(const v8bf*)(p);        // K 0..7  (or 8..15)
  u.h[1] = *(const v8bf*)(p + 16);   // K 16..23 (or 24..31)
  return u.v;
}

// B fragment (32x16 bf16): LDS stored column-major: [col][k], row stride LDT
__device__ __forceinline__ v16bf load_b(const __bf16* Bm, int n0, int lane) {
  int col = n0 + (lane & 15);
  const __bf16* p = Bm + col * LDT + ((lane & 16) ? 8 : 0);
  union { v16bf v; v8bf h[2]; } u;
  u.h[0] = *(const v8bf*)(p);        // K 0..7  (or 8..15)
  u.h[1] = *(const v8bf*)(p + 16);   // K 16..23 (or 24..31)
  return u.v;
}

// ---------------- prep: reorder weights to bf16, K = k*Ci + c ----------------
__global__ void k_prep(const float* __restrict__ wdcn, const float* __restrict__ woff,
                       __bf16* __restrict__ wb, __bf16* __restrict__ wo) {
  int i = blockIdx.x * 256 + threadIdx.x;
  if (i < Cco * KT) {
    int o = i / KT; int r = i - o * KT; int k = r >> 8; int c = r & 255;
    wb[i] = (__bf16)wdcn[(o * Ci + c) * 9 + k];
  }
  int j = i - Cco * KT;
  if (j >= 0 && j < 32 * KT) {
    int o = j / KT; int r = j - o * KT; int k = r >> 8; int c = r & 255;
    wo[j] = (o < 27) ? (__bf16)woff[(o * Ci + c) * 9 + k] : (__bf16)0.f;
  }
}

// ---------------- offset conv: WMMA implicit-GEMM 3x3, 256 -> 27(32) --------
__global__ void k_offconv(const float* __restrict__ x, const __bf16* __restrict__ wo,
                          const float* __restrict__ boff, float* __restrict__ om) {
  __shared__ __align__(16) __bf16 sA[2][32 * LDT];
  __shared__ __align__(16) __bf16 sB[2][64 * LDT];
  int tid = threadIdx.x;
  int wave = tid >> 5, lane = tid & 31;
  int pixT = blockIdx.x;
  int b = pixT / 144; int r = pixT - b * 144;
  int h0 = (r / 12) * 8, w0 = (r % 12) * 8;
  int m0 = (wave & 1) * 16;
  int n0 = (wave >> 1) * 16;
  v8f acc = {};

  auto buildA = [&](int it, int buf) {
    if (tid < 128) {  // 32 rows x 32 K = 128 x 16B
      int k = it >> 3, cb = it & 7;
      int row = tid >> 2, g = tid & 3;
      cp16(wo + (size_t)row * KT + k * 256 + cb * 32 + g * 8,
           &sA[buf][row * LDT + g * 8]);
    }
  };
  auto buildB = [&](int it, int buf) {
    int k = it >> 3, cb = it & 7;
    int ky = k / 3, kx = k - ky * 3;
    for (int s = tid; s < 1024; s += 256) {  // 64 px x 16 channel-pairs
      int p = s >> 4, cp = s & 15;
      int c0 = cb * 32 + cp * 2;
      int h = h0 + (p >> 3), w = w0 + (p & 7);
      int ys = h + ky - 1, xs = w + kx - 1;
      bool ok = ((unsigned)ys < 96u) && ((unsigned)xs < 96u);
      int sp = ok ? (ys * Wd + xs) : 0;
      const float* xb = x + (size_t)(b * Ci + c0) * HW;
      float v0 = ok ? xb[sp] : 0.f;
      float v1 = ok ? xb[sp + HW] : 0.f;
      *(unsigned*)&sB[buf][p * LDT + cp * 2] = pack2bf(v0, v1);
    }
  };

  buildA(0, 0);
  buildB(0, 0);
  async_wait();
  __syncthreads();
  for (int it = 0; it < 72; ++it) {
    int cur = it & 1;
    if (it + 1 < 72) {  // overlap next tile build with this tile's WMMA
      buildA(it + 1, cur ^ 1);
      buildB(it + 1, cur ^ 1);
    }
    v16bf af = load_a(sA[cur], m0, lane);
    v16bf bf = load_b(sB[cur], n0, lane);
    acc = wmma_bf16(af, bf, acc);
    async_wait();
    __syncthreads();
  }

  int colp = n0 + (lane & 15);
  int rb = m0 + ((lane & 16) ? 8 : 0);
  int h = h0 + (colp >> 3), w = w0 + (colp & 7);
#pragma unroll
  for (int rr = 0; rr < 8; ++rr) {
    int o = rb + rr;
    if (o < 27) om[(size_t)(b * 27 + o) * HW + h * Wd + w] = acc[rr] + boff[o];
  }
}

// ---------------- main fused deformable conv -------------------------------
__global__ void k_dcn(const float* __restrict__ x, const __bf16* __restrict__ wb,
                      const float* __restrict__ om, const float* __restrict__ bdcn,
                      float* __restrict__ pre, float* __restrict__ partS,
                      float* __restrict__ partQ) {
  __shared__ __align__(16) __bf16 sA[2][64 * LDT];
  __shared__ __align__(16) __bf16 sB[2][64 * LDT];
  __shared__ int   sIdx[576 * 4];
  __shared__ float sWt [576 * 4];
  __shared__ float sS[64 * 4];
  __shared__ float sQ[64 * 4];

  int tid = threadIdx.x;
  int wave = tid >> 5, lane = tid & 31;
  int bid = blockIdx.x;
  int pixT = bid % PIXT;
  int coBase = (bid / PIXT) * 64;
  int b = pixT / 144; int r = pixT - b * 144;
  int h0 = (r / 12) * 8, w0 = (r % 12) * 8;

  // ---- sampling metadata: 64 pixels x 9 taps -> 4 (idx, weight*mask) each
  for (int i = tid; i < 576; i += 256) {
    int p = i / 9, k = i - p * 9;
    int h = h0 + (p >> 3), w = w0 + (p & 7);
    int ky = k / 3, kx = k - ky * 3;
    size_t ob = (size_t)(b * 27) * HW + h * Wd + w;
    float dy = om[ob + (size_t)(2 * k) * HW];
    float dx = om[ob + (size_t)(2 * k + 1) * HW];
    float mk = om[ob + (size_t)(18 + k) * HW];
    float msk = 1.f / (1.f + __expf(-mk));
    float py = dy + (float)(h - 1 + ky);
    float px = dx + (float)(w - 1 + kx);
    float y0f = floorf(py), x0f = floorf(px);
    float wy1 = py - y0f, wx1 = px - x0f;
    float wy[2] = {1.f - wy1, wy1};
    float wx[2] = {1.f - wx1, wx1};
    int y0 = (int)y0f, x0 = (int)x0f;
#pragma unroll
    for (int j = 0; j < 4; ++j) {
      int yy = y0 + (j >> 1), xx = x0 + (j & 1);
      bool ok = ((unsigned)yy < 96u) && ((unsigned)xx < 96u);
      sIdx[i * 4 + j] = ok ? (yy * Wd + xx) : 0;
      sWt [i * 4 + j] = ok ? (wy[j >> 1] * wx[j & 1] * msk) : 0.f;
    }
  }

  int m0 = (wave >> 1) * 16;
  int nc = wave & 1;
  int n00 = nc * 32, n01 = nc * 32 + 16;
  v8f acc0 = {}, acc1 = {};

  auto buildA = [&](int it, int buf) {  // 64 rows x 32 K = 256 x 16B
    int k = it >> 3, cb = it & 7;
    int row = tid >> 2, g = tid & 3;
    cp16(wb + (size_t)(coBase + row) * KT + k * 256 + cb * 32 + g * 8,
         &sA[buf][row * LDT + g * 8]);
  };
  auto buildB = [&](int it, int buf) {  // 64 px x 32 c bilinear-sampled x
    int k = it >> 3, cb = it & 7;
    for (int s = tid; s < 1024; s += 256) {
      int p = s >> 4, cp = s & 15;
      int c0 = cb * 32 + cp * 2;
      int mi = (p * 9 + k) * 4;
      int i0 = sIdx[mi], i1 = sIdx[mi + 1], i2 = sIdx[mi + 2], i3 = sIdx[mi + 3];
      float w0f = sWt[mi], w1f = sWt[mi + 1], w2f = sWt[mi + 2], w3f = sWt[mi + 3];
      const float* xb = x + (size_t)(b * Ci + c0) * HW;
      float v0 = w0f * xb[i0] + w1f * xb[i1] + w2f * xb[i2] + w3f * xb[i3];
      const float* xb1 = xb + HW;
      float v1 = w0f * xb1[i0] + w1f * xb1[i1] + w2f * xb1[i2] + w3f * xb1[i3];
      *(unsigned*)&sB[buf][p * LDT + cp * 2] = pack2bf(v0, v1);
    }
  };

  __syncthreads();  // meta visible to all
  buildA(0, 0);
  buildB(0, 0);
  async_wait();
  __syncthreads();
  for (int it = 0; it < 72; ++it) {
    int cur = it & 1;
    if (it + 1 < 72) {  // overlap next tile build with this tile's WMMA
      buildA(it + 1, cur ^ 1);
      buildB(it + 1, cur ^ 1);
    }
    v16bf af = load_a(sA[cur], m0, lane);
    v16bf b0 = load_b(sB[cur], n00, lane);
    v16bf b1 = load_b(sB[cur], n01, lane);
    acc0 = wmma_bf16(af, b0, acc0);
    acc1 = wmma_bf16(af, b1, acc1);
    async_wait();
    __syncthreads();
  }

  // ---- epilogue: bias, pre-BN store, deterministic per-channel partials
  int rb = m0 + ((lane & 16) ? 8 : 0);
#pragma unroll
  for (int t = 0; t < 2; ++t) {
    int n0 = t ? n01 : n00;
    v8f a = t ? acc1 : acc0;
    int colp = n0 + (lane & 15);
    int h = h0 + (colp >> 3), w = w0 + (colp & 7);
    int cg = n0 >> 4;
#pragma unroll
    for (int rr = 0; rr < 8; ++rr) {
      int o = coBase + rb + rr;
      float v = a[rr] + bdcn[o];
      pre[(size_t)(b * Cco + o) * HW + h * Wd + w] = v;
      float sv = v, qv = v * v;
#pragma unroll
      for (int m = 1; m < 16; m <<= 1) {
        sv += __shfl_xor(sv, m, 32);
        qv += __shfl_xor(qv, m, 32);
      }
      if ((lane & 15) == 0) {
        sS[(rb + rr) * 4 + cg] = sv;
        sQ[(rb + rr) * 4 + cg] = qv;
      }
    }
  }
  __syncthreads();
  if (tid < 64) {
    float s = sS[tid * 4] + sS[tid * 4 + 1] + sS[tid * 4 + 2] + sS[tid * 4 + 3];
    float q = sQ[tid * 4] + sQ[tid * 4 + 1] + sQ[tid * 4 + 2] + sQ[tid * 4 + 3];
    partS[(size_t)(coBase + tid) * PIXT + pixT] = s;
    partQ[(size_t)(coBase + tid) * PIXT + pixT] = q;
  }
}

// ---------------- BN stats (fixed-order, deterministic) --------------------
__global__ void k_stats(const float* __restrict__ partS, const float* __restrict__ partQ,
                        float* __restrict__ stats) {
  int o = threadIdx.x;  // 256 threads
  float s = 0.f, q = 0.f;
  for (int t = 0; t < PIXT; ++t) {
    s += partS[o * PIXT + t];
    q += partQ[o * PIXT + t];
  }
  float n = (float)NPIX;
  float mean = s / n;
  float var = q / n - mean * mean;
  stats[o]       = mean;
  stats[256 + o] = rsqrtf(var + 1e-5f);
}

// ---------------- BN apply + ReLU ------------------------------------------
__global__ void k_bnrelu(const float* __restrict__ pre, const float* __restrict__ stats,
                         const float* __restrict__ gamma, const float* __restrict__ beta,
                         float* __restrict__ out) {
  size_t i = (size_t)blockIdx.x * 256 + threadIdx.x;
  int o = (int)((i / HW) & 255);
  float v = (pre[i] - stats[o]) * stats[256 + o] * gamma[o] + beta[o];
  out[i] = v > 0.f ? v : 0.f;
}

extern "C" void kernel_launch(void* const* d_in, const int* in_sizes, int n_in,
                              void* d_out, int out_size, void* d_ws, size_t ws_size,
                              hipStream_t stream) {
  (void)in_sizes; (void)n_in; (void)out_size; (void)ws_size;
  const float* x     = (const float*)d_in[0];
  const float* woff  = (const float*)d_in[1];
  const float* boff  = (const float*)d_in[2];
  const float* wdcn  = (const float*)d_in[3];
  const float* bdcn  = (const float*)d_in[4];
  const float* gamma = (const float*)d_in[5];
  const float* beta  = (const float*)d_in[6];
  char* ws = (char*)d_ws;
  __bf16* wb = (__bf16*)(ws + WS_WB);
  __bf16* wo = (__bf16*)(ws + WS_WO);
  float* om  = (float*)(ws + WS_OM);
  float* pre = (float*)(ws + WS_PRE);
  float* pS  = (float*)(ws + WS_PS);
  float* pQ  = (float*)(ws + WS_PQ);
  float* st  = (float*)(ws + WS_ST);
  float* out = (float*)d_out;

  hipLaunchKernelGGL(k_prep,    dim3(2592),  dim3(256), 0, stream, wdcn, woff, wb, wo);
  hipLaunchKernelGGL(k_offconv, dim3(288),   dim3(256), 0, stream, x, wo, boff, om);
  hipLaunchKernelGGL(k_dcn,     dim3(1152),  dim3(256), 0, stream, x, wb, om, bdcn, pre, pS, pQ);
  hipLaunchKernelGGL(k_stats,   dim3(1),     dim3(256), 0, stream, pS, pQ, st);
  hipLaunchKernelGGL(k_bnrelu,  dim3(18432), dim3(256), 0, stream, pre, st, gamma, beta, out);
}
